// SCLayer_29343216566818
// MI455X (gfx1250) — matrix operand order repaired and checked by reading, not compiled, not run on hardware
//
#include <hip/hip_runtime.h>
#include <hip/hip_bf16.h>
#include <math.h>

// ---------------- problem constants ----------------
#define B_  4
#define C_  32
#define N_  256
#define T_  288
#define NP_ 12          // NUM_PRED
#define NL_ 12          // NUM_LOCAL (conv taps)
#define CIN_ 416        // 13*C
#define TPAD_ 301       // T + NP + 1 (left zero pad)
#define TOUT_ 290       // 301 - 12 + 1
#define KCH_ 156        // 4992 / 32 K-chunks
#define MT_  4          // t-subtiles (16 each) per wave in main GEMM
#define WR_ELE (2*2*156*32*16)   // conv x ctile x q x lane x halfs

typedef _Float16 h16;
typedef __attribute__((ext_vector_type(16))) _Float16 v16h;
typedef __attribute__((ext_vector_type(8)))  _Float16 v8h;
typedef __attribute__((ext_vector_type(8)))  float    v8f;

union V16 { v16h v; v8h h[2]; };

// ---------------- workspace layout (bytes) ----------------
static const size_t XC_BYTES    = (size_t)B_*N_*TPAD_*CIN_*sizeof(h16);   // 256.4 MB f16 x_cat (padded)
static const size_t ATAIL_BYTES = (size_t)B_*N_*32*CIN_*sizeof(h16);      // aux tail, tau 274..305
static const size_t X3_BYTES    = (size_t)B_*C_*N_*T_*sizeof(float);      // x3 for spatial norm
static const size_t TAILS_BYTES = (size_t)4*B_*C_*N_*5*sizeof(float);     // last-5 of x1..x4
static const size_t ADJ_BYTES   = (size_t)N_*N_*sizeof(float);
static const size_t WR_BYTES    = (size_t)WR_ELE*sizeof(h16);
static const size_t PROD_BYTES  = (size_t)B_*N_*TOUT_*C_*sizeof(float);   // g1*g2
static const size_t AUXP_BYTES  = (size_t)B_*N_*12*C_*sizeof(float);      // (a1*a2) tail

// =====================================================================
// adjacency: softmax(emb @ emb^T - 10 I) row-wise
// =====================================================================
__global__ __launch_bounds__(256) void adj_kernel(const float* __restrict__ emb,
                                                  float* __restrict__ adj) {
    int n = blockIdx.x, m = threadIdx.x;
    __shared__ float en[16];
    __shared__ float red[256];
    if (m < 16) en[m] = emb[(size_t)n*16 + m];
    __syncthreads();
    float s = 0.f;
    #pragma unroll
    for (int e = 0; e < 16; ++e) s += en[e] * emb[(size_t)m*16 + e];
    if (m == n) s -= 10.f;
    red[m] = s; __syncthreads();
    for (int o = 128; o > 0; o >>= 1) { if (m < o) red[m] = fmaxf(red[m], red[m+o]); __syncthreads(); }
    float mx = red[0]; __syncthreads();
    float e = __expf(s - mx);
    red[m] = e; __syncthreads();
    for (int o = 128; o > 0; o >>= 1) { if (m < o) red[m] += red[m+o]; __syncthreads(); }
    adj[(size_t)n*N_ + m] = e / red[0];
}

// =====================================================================
// stage1: per-(b,c,n) row — term_norm(48), seasonal_norm(24), term_norm(12)
// writes x_cat blocks 0..9 (minus RE tails), aux-tail blocks, x3 buffer, tails
// =====================================================================
__global__ __launch_bounds__(288) void stage1_kernel(const float* __restrict__ x,
                                                     h16* __restrict__ Xc,
                                                     h16* __restrict__ atail,
                                                     float* __restrict__ x3buf,
                                                     float* __restrict__ tails) {
    int n = blockIdx.x, c = blockIdx.y, b = blockIdx.z;
    int t = threadIdx.x;                     // 0..287
    __shared__ float xs[288], x1s[288], x2s[288];
    __shared__ float m1s[288], s1s[288], m3s[288], s3s[288];
    __shared__ float m2s[24], s2s[24];

    const size_t rowOff = ((size_t)(b*C_ + c)*N_ + n) * T_;
    xs[t] = x[rowOff + t];
    __syncthreads();

    // long-term norm, L = 48
    {
        int tt = t < 47 ? 47 : t;
        float s = 0.f, s2 = 0.f;
        for (int j = tt - 47; j <= tt; ++j) { float v = xs[j]; s += v; s2 += v*v; }
        float mean = s * (1.f/48.f);
        float var  = s2 * (1.f/48.f) - mean*mean + 1e-5f;
        m1s[t] = mean; s1s[t] = sqrtf(var);
        x1s[t] = (xs[t] - mean) * rsqrtf(var + 1.0f);
    }
    __syncthreads();

    // seasonal norm, P = 24 (12 repeats)
    if (t < 24) {
        float s = 0.f, s2 = 0.f;
        for (int r = 0; r < 12; ++r) { float v = x1s[t + 24*r]; s += v; s2 += v*v; }
        float mean = s * (1.f/12.f);
        float var  = s2 * (1.f/12.f) - mean*mean + 1e-5f;
        m2s[t] = mean; s2s[t] = sqrtf(var);
    }
    __syncthreads();
    {
        int p = t % 24;
        float std2 = s2s[p];
        x2s[t] = (x1s[t] - m2s[p]) * rsqrtf(std2*std2 + 1.0f);
    }
    __syncthreads();

    const size_t xcb = ((size_t)(b*N_ + n)) * TPAD_ * CIN_ + c;

    // short-term norm, L = 12 (on x2)
    {
        int tt = t < 11 ? 11 : t;
        float s = 0.f, s2 = 0.f;
        for (int j = tt - 11; j <= tt; ++j) { float v = x2s[j]; s += v; s2 += v*v; }
        float mean = s * (1.f/12.f);
        float var  = s2 * (1.f/12.f) - mean*mean + 1e-5f;
        m3s[t] = mean; s3s[t] = sqrtf(var);
        float x3 = (x2s[t] - mean) * rsqrtf(var + 1.0f);
        x3buf[rowOff + t] = x3;
        Xc[xcb + (size_t)(t+1)*CIN_ + 7*32] = (h16)x3;   // block 7 (xp3), tau 1..288
        if (t >= 283) tails[((((size_t)2*B_ + b)*C_ + c)*N_ + n)*5 + (t-283)] = x3;
    }
    if (t >= 283) {
        tails[((((size_t)0*B_ + b)*C_ + c)*N_ + n)*5 + (t-283)] = x1s[t];
        tails[((((size_t)1*B_ + b)*C_ + c)*N_ + n)*5 + (t-283)] = x2s[t];
    }
    __syncthreads();

    // tau = 0 zero pad for this channel in all 13 blocks
    if (t == 0) {
        #pragma unroll
        for (int blk = 0; blk < 13; ++blk) Xc[xcb + (size_t)blk*32] = (h16)0.f;
    }

    // x_cat write phase, tt = 0..299 -> tau = tt+1
    for (int tt = t; tt < 300; tt += 288) {
        int tc = tt < 287 ? tt : 287;
        size_t base = xcb + (size_t)(tt + 1) * CIN_;
        Xc[base + 0*32] = (h16)xs[tc];                         // x_proj (const extrap)
        if (tt < 288) {
            Xc[base + 1*32] = (h16)x1s[tt];                    // xp1 body
            Xc[base + 4*32] = (h16)x2s[tt];                    // xp2 body
        }
        Xc[base + 2*32] = (h16)m1s[tc];
        Xc[base + 3*32] = (h16)s1s[tc];
        Xc[base + 5*32] = (h16)m2s[tt % 24];                   // seasonal extrap == periodic
        Xc[base + 6*32] = (h16)s2s[tt % 24];
        Xc[base + 8*32] = (h16)m3s[tc];
        Xc[base + 9*32] = (h16)s3s[tc];
    }

    // aux tail (tau = 274 + tloc), zero blocks 0,1,4,7,10; data blocks 2,3,5,6,8,9
    if (t < 32) {
        int tt = 273 + t;
        int tc = tt < 287 ? tt : 287;
        size_t ab = ((size_t)(b*N_ + n)*32 + t) * CIN_ + c;
        atail[ab + 0*32]  = (h16)0.f;
        atail[ab + 1*32]  = (h16)0.f;
        atail[ab + 4*32]  = (h16)0.f;
        atail[ab + 7*32]  = (h16)0.f;
        atail[ab + 10*32] = (h16)0.f;
        atail[ab + 2*32]  = (h16)m1s[tc];
        atail[ab + 3*32]  = (h16)s1s[tc];
        atail[ab + 5*32]  = (h16)m2s[tt % 24];
        atail[ab + 6*32]  = (h16)s2s[tt % 24];
        atail[ab + 8*32]  = (h16)m3s[tc];
        atail[ab + 9*32]  = (h16)s3s[tc];
    }
}

// =====================================================================
// spatial norm: mean/var via adj row GEMV; writes blocks 10,11,12
// =====================================================================
__global__ __launch_bounds__(288) void spatial_kernel(const float* __restrict__ x3buf,
                                                      const float* __restrict__ adj,
                                                      h16* __restrict__ Xc,
                                                      h16* __restrict__ atail,
                                                      float* __restrict__ tails) {
    int n = blockIdx.x, c = blockIdx.y, b = blockIdx.z;
    int t = threadIdx.x;
    __shared__ float m4s[288], s4s[288];
    const float* xb   = x3buf + ((size_t)(b*C_ + c)*N_) * T_;
    const float* arow = adj + (size_t)n * N_;
    float mean = 0.f, mm = 0.f;
    for (int m = 0; m < N_; ++m) {
        float a = arow[m];
        float v = xb[(size_t)m*T_ + t];
        mean += a * v;
        mm   += a * v * v;
    }
    float var  = mm - mean*mean + 1e-5f;
    float stdv = sqrtf(var);
    float x4 = (xb[(size_t)n*T_ + t] - mean) * rsqrtf(var + 1.0f);
    m4s[t] = mean; s4s[t] = stdv;
    size_t xcb = ((size_t)(b*N_ + n)) * TPAD_ * CIN_ + c;
    Xc[xcb + (size_t)(t+1)*CIN_ + 10*32] = (h16)x4;
    if (t >= 283) tails[((((size_t)3*B_ + b)*C_ + c)*N_ + n)*5 + (t-283)] = x4;
    __syncthreads();
    for (int tt = t; tt < 300; tt += 288) {
        int tc = tt < 287 ? tt : 287;
        size_t base = xcb + (size_t)(tt + 1) * CIN_;
        Xc[base + 11*32] = (h16)m4s[tc];
        Xc[base + 12*32] = (h16)s4s[tc];
    }
    if (t < 32) {
        int tt = 273 + t;
        int tc = tt < 287 ? tt : 287;
        size_t ab = ((size_t)(b*N_ + n)*32 + t) * CIN_ + c;
        atail[ab + 11*32] = (h16)m4s[tc];
        atail[ab + 12*32] = (h16)s4s[tc];
    }
}

// =====================================================================
// residual extrapolation: proj[b,c,n,p] from last 5 steps; fills tau 289..300
// of blocks 1 (x1), 4 (x2), 7 (x3), 10 (x4)
// =====================================================================
__global__ __launch_bounds__(384) void re_kernel(const float* __restrict__ tails,
        const float* __restrict__ w0, const float* __restrict__ b0,
        const float* __restrict__ w1, const float* __restrict__ b1,
        const float* __restrict__ w2, const float* __restrict__ b2,
        const float* __restrict__ w3, const float* __restrict__ b3,
        h16* __restrict__ Xc) {
    int i = blockIdx.x, n = blockIdx.y, b = blockIdx.z;
    const float* W    = (i == 0) ? w0 : (i == 1) ? w1 : (i == 2) ? w2 : w3;
    const float* bias = (i == 0) ? b0 : (i == 1) ? b1 : (i == 2) ? b2 : b3;
    __shared__ float tl[160];                 // [ci][k]
    int tid = threadIdx.x;
    if (tid < 160) {
        int c = tid / 5, k = tid % 5;
        tl[tid] = tails[((((size_t)i*B_ + b)*C_ + c)*N_ + n)*5 + k];
    }
    __syncthreads();
    int p  = tid >> 5;
    int co = tid & 31;
    int o  = p * C_ + co;
    float v = bias[o];
    const float* wr = W + (size_t)o * 160;    // W[o][ci][0][k] flat
    #pragma unroll 5
    for (int j = 0; j < 160; ++j) v += wr[j] * tl[j];
    int blk = 3*i + 1;
    Xc[((size_t)(b*N_ + n)*TPAD_ + (289 + p)) * CIN_ + blk*32 + co] = (h16)v;
}

// =====================================================================
// conv weight repack into WMMA B-fragment layout:
// Wr[conv][ctile][q][lane][h] = W[co][ci][k],
//   co = ctile*16 + (lane&15), j = (lane>>4)*8 + (h<8 ? h : h+8), k = q/13, ci = (q%13)*32 + j
// =====================================================================
__global__ __launch_bounds__(256) void repack_kernel(const float* __restrict__ w1,
                                                     const float* __restrict__ w2,
                                                     h16* __restrict__ Wr) {
    int idx = blockIdx.x * 256 + threadIdx.x;
    if (idx >= WR_ELE) return;
    int h    = idx & 15;
    int lane = (idx >> 4) & 31;
    int q    = (idx >> 9) % KCH_;
    int ct   = (idx / (KCH_ * 512)) & 1;
    int conv = idx / (2 * KCH_ * 512);
    int co = ct*16 + (lane & 15);
    int j  = ((lane >> 4) << 3) + (h < 8 ? h : h + 8);
    int k  = q / 13;
    int ci = (q % 13) * 32 + j;
    const float* W = conv ? w2 : w1;
    Wr[idx] = (h16)W[((size_t)co*CIN_ + ci)*NL_ + k];
}

// =====================================================================
// main conv-as-GEMM with WMMA f16->f32, fused g1*g2.
// One wave per (b, n, 64-t block): MT_=4 sub-tiles of M=16 share every
// B fragment -> 16 WMMA per 16 vmem loads per K-chunk.
// K = 156 chunks of 32 (416 = 13*32, chunks never straddle a conv tap).
// =====================================================================
__global__ __launch_bounds__(32) void conv_wmma_kernel(const h16* __restrict__ Xc,
                                                       const h16* __restrict__ Wr,
                                                       const float* __restrict__ bias1,
                                                       const float* __restrict__ bias2,
                                                       float* __restrict__ prod) {
    int blk = blockIdx.x;                  // 0..4 (covers t 0..319, tail clamped)
    int n = blockIdx.y, b = blockIdx.z;
    int lane = threadIdx.x;
    int mrow = lane & 15, grp = lane >> 4;
    int t0 = blk * (16 * MT_);
    size_t bn = (size_t)b * N_ + n;
    const h16* Xrow = Xc + bn * TPAD_ * CIN_;
    const size_t ctStride = (size_t)KCH_ * 512;

    v8f acc[MT_][2][2] = {};
    for (int q = 0; q < KCH_; ++q) {
        int k  = q / 13;
        int c0 = (q - k*13) * 32;
        const h16* pb = Wr + ((size_t)q * 32 + lane) * 16;
        v16h b00 = *(const v16h*)(pb);                   // conv1, co 0..15
        v16h b01 = *(const v16h*)(pb + ctStride);        // conv1, co 16..31
        v16h b10 = *(const v16h*)(pb + 2*ctStride);      // conv2, co 0..15
        v16h b11 = *(const v16h*)(pb + 3*ctStride);      // conv2, co 16..31
        V16 A[MT_];
        #pragma unroll
        for (int s = 0; s < MT_; ++s) {
            int tau = t0 + s*16 + mrow + k;
            if (tau > 300) tau = 300;                    // safe clamp (rows t>=290 discarded)
            const h16* pa = Xrow + (size_t)tau * CIN_ + c0 + grp*8;
            A[s].h[0] = *(const v8h*)(pa);               // K = grp*8 + 0..7
            A[s].h[1] = *(const v8h*)(pa + 16);          // K = 16 + grp*8 + 0..7
        }
        #pragma unroll
        for (int s = 0; s < MT_; ++s) {
            acc[s][0][0] = __builtin_amdgcn_wmma_f32_16x16x32_f16(false, A[s].v, false, b00, (short)0, acc[s][0][0], false, false);
            acc[s][0][1] = __builtin_amdgcn_wmma_f32_16x16x32_f16(false, A[s].v, false, b01, (short)0, acc[s][0][1], false, false);
            acc[s][1][0] = __builtin_amdgcn_wmma_f32_16x16x32_f16(false, A[s].v, false, b10, (short)0, acc[s][1][0], false, false);
            acc[s][1][1] = __builtin_amdgcn_wmma_f32_16x16x32_f16(false, A[s].v, false, b11, (short)0, acc[s][1][1], false, false);
        }
    }
    #pragma unroll
    for (int ct = 0; ct < 2; ++ct) {
        int co = ct*16 + (lane & 15);
        float bb1 = bias1[co], bb2 = bias2[co];
        #pragma unroll
        for (int s = 0; s < MT_; ++s) {
            #pragma unroll
            for (int r = 0; r < 8; ++r) {
                int t = t0 + s*16 + r + 8*grp;           // C/D layout: lanes 16-31 hold M+8
                if (t < TOUT_) {
                    float pv = (acc[s][0][ct][r] + bb1) * (acc[s][1][ct][r] + bb2);
                    prod[(bn * TOUT_ + t) * C_ + co] = pv;
                }
            }
        }
    }
}

// aux variant: a_cat tail only (t = 274..289 -> keep 278..289)
__global__ __launch_bounds__(32) void conv_wmma_aux_kernel(const h16* __restrict__ atail,
                                                           const h16* __restrict__ Wr,
                                                           const float* __restrict__ bias1,
                                                           const float* __restrict__ bias2,
                                                           float* __restrict__ auxp) {
    int n = blockIdx.x, b = blockIdx.y;
    int lane = threadIdx.x;
    int mrow = lane & 15, grp = lane >> 4;
    size_t bn = (size_t)b * N_ + n;
    const h16* Xrow = atail + bn * 32 * CIN_;
    const size_t ctStride = (size_t)KCH_ * 512;

    v8f acc[2][2] = {};
    for (int q = 0; q < KCH_; ++q) {
        int k  = q / 13;
        int c0 = (q - k*13) * 32;
        int tloc = mrow + k;                 // tau - 274, max 26 < 32
        const h16* pa = Xrow + (size_t)tloc * CIN_ + c0 + grp*8;
        V16 A;
        A.h[0] = *(const v8h*)(pa);
        A.h[1] = *(const v8h*)(pa + 16);
        const h16* pb = Wr + ((size_t)q * 32 + lane) * 16;
        v16h b00 = *(const v16h*)(pb);
        v16h b01 = *(const v16h*)(pb + ctStride);
        v16h b10 = *(const v16h*)(pb + 2*ctStride);
        v16h b11 = *(const v16h*)(pb + 3*ctStride);
        acc[0][0] = __builtin_amdgcn_wmma_f32_16x16x32_f16(false, A.v, false, b00, (short)0, acc[0][0], false, false);
        acc[0][1] = __builtin_amdgcn_wmma_f32_16x16x32_f16(false, A.v, false, b01, (short)0, acc[0][1], false, false);
        acc[1][0] = __builtin_amdgcn_wmma_f32_16x16x32_f16(false, A.v, false, b10, (short)0, acc[1][0], false, false);
        acc[1][1] = __builtin_amdgcn_wmma_f32_16x16x32_f16(false, A.v, false, b11, (short)0, acc[1][1], false, false);
    }
    #pragma unroll
    for (int ct = 0; ct < 2; ++ct) {
        int co = ct*16 + (lane & 15);
        float bb1 = bias1[co], bb2 = bias2[co];
        #pragma unroll
        for (int r = 0; r < 8; ++r) {
            int t = 274 + r + 8*grp;
            int p = t - 278;
            if (p >= 0 && p < 12) {
                float pv = (acc[0][ct][r] + bb1) * (acc[1][ct][r] + bb2);
                auxp[(bn * 12 + p) * C_ + co] = pv;
            }
        }
    }
}

// =====================================================================
// 1x1 conv over channel dim; dst laid out (b, co, n, tCount) flat
// =====================================================================
__global__ __launch_bounds__(256) void conv1x1_kernel(const float* __restrict__ src,
                                                      int Tsrc, int tStart, int tCount,
                                                      const float* __restrict__ W,
                                                      const float* __restrict__ bias,
                                                      float* __restrict__ dst) {
    size_t idx = (size_t)blockIdx.x * 256 + threadIdx.x;
    size_t total = (size_t)B_ * C_ * N_ * tCount;
    if (idx >= total) return;
    int t = (int)(idx % tCount); size_t r = idx / tCount;
    int n = (int)(r % N_); r /= N_;
    int co = (int)(r % C_);
    int b  = (int)(r / C_);
    const float* row = src + (((size_t)(b*N_ + n)) * Tsrc + (tStart + t)) * C_;
    const float* w = W + (size_t)co * C_;
    float v = bias[co];
    #pragma unroll
    for (int c = 0; c < C_; ++c) v += w[c] * row[c];
    dst[idx] = v;
}

// =====================================================================
extern "C" void kernel_launch(void* const* d_in, const int* in_sizes, int n_in,
                              void* d_out, int out_size, void* d_ws, size_t ws_size,
                              hipStream_t stream) {
    const float* x       = (const float*)d_in[0];
    const float* emb     = (const float*)d_in[1];
    const float* conv1_w = (const float*)d_in[2];
    const float* conv1_b = (const float*)d_in[3];
    const float* conv2_w = (const float*)d_in[4];
    const float* conv2_b = (const float*)d_in[5];
    const float* skip_w  = (const float*)d_in[6];
    const float* skip_b  = (const float*)d_in[7];
    const float* res_w   = (const float*)d_in[8];
    const float* res_b   = (const float*)d_in[9];
    const float* re_w[4] = { (const float*)d_in[10], (const float*)d_in[12],
                             (const float*)d_in[14], (const float*)d_in[16] };
    const float* re_b[4] = { (const float*)d_in[11], (const float*)d_in[13],
                             (const float*)d_in[15], (const float*)d_in[17] };

    char* ws = (char*)d_ws;
    size_t off = 0;
    h16*   Xc    = (h16*)  (ws + off); off += XC_BYTES;
    h16*   atail = (h16*)  (ws + off); off += ATAIL_BYTES;
    float* x3buf = (float*)(ws + off); off += X3_BYTES;
    float* tails = (float*)(ws + off); off += TAILS_BYTES;
    float* adj   = (float*)(ws + off); off += ADJ_BYTES;
    h16*   Wr    = (h16*)  (ws + off); off += WR_BYTES;
    float* prod  = (float*)(ws + off); off += PROD_BYTES;
    float* auxp  = (float*)(ws + off); off += AUXP_BYTES;

    float* out   = (float*)d_out;
    float* out_xz   = out;                                // (4,32,256,278)
    float* out_saux = out + (size_t)B_*C_*N_*278;         // (4,32,256,12)
    float* out_s    = out_saux + (size_t)B_*C_*N_*12;     // (4,32,256,12)

    adj_kernel<<<N_, 256, 0, stream>>>(emb, adj);
    stage1_kernel<<<dim3(N_, C_, B_), 288, 0, stream>>>(x, Xc, atail, x3buf, tails);
    spatial_kernel<<<dim3(N_, C_, B_), 288, 0, stream>>>(x3buf, adj, Xc, atail, tails);
    re_kernel<<<dim3(4, N_, B_), 384, 0, stream>>>(tails,
        re_w[0], re_b[0], re_w[1], re_b[1], re_w[2], re_b[2], re_w[3], re_b[3], Xc);
    repack_kernel<<<(WR_ELE + 255) / 256, 256, 0, stream>>>(conv1_w, conv2_w, Wr);

    conv_wmma_kernel<<<dim3(5, N_, B_), 32, 0, stream>>>(Xc, Wr, conv1_b, conv2_b, prod);
    conv_wmma_aux_kernel<<<dim3(N_, B_), 32, 0, stream>>>(atail, Wr, conv1_b, conv2_b, auxp);

    {   // x_z = res_w * prod[:, :278]
        size_t total = (size_t)B_*C_*N_*278;
        conv1x1_kernel<<<(unsigned)((total + 255) / 256), 256, 0, stream>>>(
            prod, TOUT_, 0, 278, res_w, res_b, out_xz);
    }
    {   // s_aux = skip_w * auxprod
        size_t total = (size_t)B_*C_*N_*12;
        conv1x1_kernel<<<(unsigned)((total + 255) / 256), 256, 0, stream>>>(
            auxp, 12, 0, 12, skip_w, skip_b, out_saux);
    }
    {   // s = skip_w * prod[:, 278:290]
        size_t total = (size_t)B_*C_*N_*12;
        conv1x1_kernel<<<(unsigned)((total + 255) / 256), 256, 0, stream>>>(
            prod, TOUT_, 278, 12, skip_w, skip_b, out_s);
    }
}
